// rcri_cm_52647709114403
// MI455X (gfx1250) — matrix operand order repaired and checked by reading, not compiled, not run on hardware
//
#include <hip/hip_runtime.h>
#include <hip/hip_bf16.h>
#include <math.h>

// ---------------------------------------------------------------------------
// Problem constants (from the reference)
// ---------------------------------------------------------------------------
#define B_   8
#define N_   8192
#define S_   2048
#define K_   32
#define D_   64
#define P_   (B_ * S_ * K_)      // 524288 positions = B*S*K, p = (b*S+s)*K + k
#define EPS_    1e-7f
#define BN_EPS_ 1e-5f

typedef __attribute__((ext_vector_type(16))) _Float16 v16h;
typedef __attribute__((ext_vector_type(8)))  _Float16 v8h;
typedef __attribute__((ext_vector_type(8)))  float    v8f;

// ---------------------------------------------------------------------------
// 0) init: zero BN stats, set identity transforms (layer-1 input + raw concat)
// ---------------------------------------------------------------------------
__global__ void rcri_init_kernel(float* __restrict__ stats, int nstats,
                                 float* __restrict__ t1,    // stride 32
                                 float* __restrict__ t3)    // stride 128
{
    for (int i = threadIdx.x; i < nstats; i += blockDim.x) stats[i] = 0.0f;
    int c = threadIdx.x;
    if (c < 32) {                      // layer-1 input: raw rif, no BN, no ReLU
        t1[c] = 1.0f; t1[32 + c] = 0.0f; t1[64 + c] = -INFINITY;
    }
    if (c >= 64 && c < 128) {          // layer-3 concat half: raw grouped fmaps
        t3[c] = 1.0f; t3[128 + c] = 0.0f; t3[256 + c] = -INFINITY;
    }
}

// ---------------------------------------------------------------------------
// 0b) zero-fill (b128 stores) for the padded channels of the layer-1 input
// ---------------------------------------------------------------------------
__global__ void rcri_zeropad_kernel(v8h* __restrict__ dst, long long n8)
{
    const long long i = (long long)blockIdx.x * blockDim.x + threadIdx.x;
    if (i < n8) { v8h z = 0; dst[i] = z; }
}

// ---------------------------------------------------------------------------
// 1) weight f32 -> f16, zero-padded in Cin
// ---------------------------------------------------------------------------
__global__ void rcri_wconv_kernel(const float* __restrict__ W,
                                  _Float16* __restrict__ Wh,
                                  int Cout, int Cin, int CinPad)
{
    int t = blockIdx.x * blockDim.x + threadIdx.x;
    if (t >= Cout * CinPad) return;
    int o = t / CinPad, c = t % CinPad;
    Wh[t] = (_Float16)((c < Cin) ? W[o * Cin + c] : 0.0f);
}

// ---------------------------------------------------------------------------
// 2) geometry: rif features (8 real channels of 32), + sampled/new_LOA
// ---------------------------------------------------------------------------
__device__ __forceinline__ float acos_cl(float x) {
    return acosf(fminf(fmaxf(x, -1.0f + EPS_), 1.0f - EPS_));
}

__global__ void rcri_geom_kernel(const float* __restrict__ contour,
                                 const float* __restrict__ loa,
                                 _Float16* __restrict__ rif,        // [32][P]
                                 float* __restrict__ out_sampled,   // [B][S][3]
                                 float* __restrict__ out_loa)       // [B][S][3]
{
    const int p  = blockIdx.x * blockDim.x + threadIdx.x;  // p = (b*S+s)*K + k
    const int k  = p & (K_ - 1);
    const int bs = p >> 5;
    const int s  = bs & (S_ - 1);
    const int b  = bs >> 11;

    const int ni = (int)(((long long)s * (N_ - 1)) / (S_ - 1));  // linspace idx
    const int g  = (ni - K_ / 2 + k) & (N_ - 1);
    const int gm = (ni - K_ / 2 + ((k + K_ - 1) & (K_ - 1))) & (N_ - 1);

    const size_t cb = (size_t)b * N_ * 3;
    const float psx = contour[cb + (size_t)ni * 3 + 0];
    const float psy = contour[cb + (size_t)ni * 3 + 1];
    const float psz = contour[cb + (size_t)ni * 3 + 2];
    const float axx = loa[cb + (size_t)ni * 3 + 0];
    const float axy = loa[cb + (size_t)ni * 3 + 1];
    const float axz = loa[cb + (size_t)ni * 3 + 2];

    const float ptx = contour[cb + (size_t)g * 3 + 0];
    const float pty = contour[cb + (size_t)g * 3 + 1];
    const float ptz = contour[cb + (size_t)g * 3 + 2];
    const float lox = loa[cb + (size_t)g * 3 + 0];
    const float loy = loa[cb + (size_t)g * 3 + 1];
    const float loz = loa[cb + (size_t)g * 3 + 2];

    const float pmx = contour[cb + (size_t)gm * 3 + 0];
    const float pmy = contour[cb + (size_t)gm * 3 + 1];
    const float pmz = contour[cb + (size_t)gm * 3 + 2];
    const float lmx = loa[cb + (size_t)gm * 3 + 0];
    const float lmy = loa[cb + (size_t)gm * 3 + 1];
    const float lmz = loa[cb + (size_t)gm * 3 + 2];

    // local = pt - ps ; localm = ptm - ps
    const float lx = ptx - psx, ly = pty - psy, lz = ptz - psz;
    const float mx = pmx - psx, my = pmy - psy, mz = pmz - psz;

    const float sq1 = lx * lx + ly * ly + lz * lz;
    const float d1  = (sq1 > 0.0f) ? sqrtf(sq1) : 0.0f;
    const float i1  = (d1 > 0.0f) ? (1.0f / d1) : 0.0f;
    const float u1x = lx * i1, u1y = ly * i1, u1z = lz * i1;

    const float sqm = mx * mx + my * my + mz * mz;
    const float dm  = (sqm > 0.0f) ? sqrtf(sqm) : 0.0f;
    const float im  = (dm > 0.0f) ? (1.0f / dm) : 0.0f;
    const float umx = mx * im, umy = my * im, umz = mz * im;

    const float vx = lx - mx, vy = ly - my, vz = lz - mz;   // inner_vec
    const float sqv = vx * vx + vy * vy + vz * vz;
    const float dv  = (sqv > 0.0f) ? sqrtf(sqv) : 0.0f;
    const float iv  = (dv > 0.0f) ? (1.0f / dv) : 0.0f;
    const float ux = vx * iv, uy = vy * iv, uz = vz * iv;

    const float a1 = u1x * axx + u1y * axy + u1z * axz;
    const float a2 = u1x * lox + u1y * loy + u1z * loz;
    const float a3 = acos_cl(lox * axx + loy * axy + loz * axz);
    const float a4 = ux * lox + uy * loy + uz * loz;
    const float a5 = ux * lmx + uy * lmy + uz * lmz;
    const float a6 = acos_cl(lox * lmx + loy * lmy + loz * lmz);
    const float d2 = acos_cl(u1x * umx + u1y * umy + u1z * umz);

    const size_t pp = (size_t)p, P = (size_t)P_;
    rif[0 * P + pp] = (_Float16)d1;
    rif[1 * P + pp] = (_Float16)d2;
    rif[2 * P + pp] = (_Float16)a1;
    rif[3 * P + pp] = (_Float16)a2;
    rif[4 * P + pp] = (_Float16)a3;
    rif[5 * P + pp] = (_Float16)a4;
    rif[6 * P + pp] = (_Float16)a5;
    rif[7 * P + pp] = (_Float16)a6;

    if (k == 0) {
        const size_t o = (size_t)bs * 3;
        out_sampled[o + 0] = psx; out_sampled[o + 1] = psy; out_sampled[o + 2] = psz;
        out_loa[o + 0] = axx;     out_loa[o + 1] = axy;     out_loa[o + 2] = axz;
    }
}

// ---------------------------------------------------------------------------
// 3) gather feature_maps -> rows [64..127] of layer-3 input (raw, f16)
// ---------------------------------------------------------------------------
__global__ void rcri_gather_kernel(const float* __restrict__ fm,   // [B][D][N]
                                   _Float16* __restrict__ xin3)    // [128][P]
{
    const int p = blockIdx.x * blockDim.x + threadIdx.x;
    const int d = blockIdx.y;
    const int k  = p & (K_ - 1);
    const int bs = p >> 5;
    const int s  = bs & (S_ - 1);
    const int b  = bs >> 11;
    const int ni = (int)(((long long)s * (N_ - 1)) / (S_ - 1));
    const int g  = (ni - K_ / 2 + k) & (N_ - 1);
    const float v = fm[(size_t)b * D_ * N_ + (size_t)d * N_ + g];
    xin3[(size_t)(64 + d) * P_ + p] = (_Float16)v;
}

// ---------------------------------------------------------------------------
// 4) WMMA GEMM: Y[Cout][P] = W[Cout][Cin] * transform(X[Cin][P]) + bias
//    transform(v,c) = max(v*scale[c] + shift[c], lo[c])  (fused BN+ReLU of the
//    previous layer; lo = -inf encodes "raw, no ReLU").
//    Each wave computes a 32(M) x 64(N) tile: 2 m-tiles x 4 n-tiles
//    = 8 accumulators, 8 WMMAs per K-step of 32. The B operand (and its BN
//    transform, the dominant VALU cost) is built once per n-tile and reused
//    by both m-tiles; A (weights) is loaded once per K-step per m-tile.
//    Per-channel BN stats (sum, sumsq) are reduced across the 16-lane half
//    with shfl_xor and flushed with one pair of f32 global atomics per row.
//    All loads are unconditional (layer-1 K-padding is materialized zeros) so
//    the compiler can clause them without exec-mask gymnastics.
// ---------------------------------------------------------------------------
__global__ __launch_bounds__(128) void rcri_gemm_bn_kernel(
    const _Float16* __restrict__ X,   // [Cin][P] f16
    const _Float16* __restrict__ Wh,  // [Cout][Cin] f16 (Cin padded, %32==0)
    const float* __restrict__ bias,   // [Cout]
    const float* __restrict__ tr,     // [scale Cin][shift Cin][lo Cin]
    float* __restrict__ stats,        // [sum Cout][sumsq Cout]
    _Float16* __restrict__ Y,         // [Cout][P] f16
    int Cin, int Cout)
{
    const int lane   = threadIdx.x & 31;
    const int wave   = threadIdx.x >> 5;
    const int half16 = lane >> 4;            // which 16-lane half of the wave
    const int l      = lane & 15;
    const int p0     = (blockIdx.x * 4 + wave) * 64;   // 64-position tile
    const int m0     = blockIdx.y * 32;                // 32-out-channel tile

    const float* tsc = tr;
    const float* tsh = tr + Cin;
    const float* tlo = tr + 2 * Cin;

    v8f acc[2][4];
#pragma unroll
    for (int u = 0; u < 2; ++u)
#pragma unroll
        for (int t = 0; t < 4; ++t) acc[u][t] = 0;

    for (int kk = 0; kk < Cin; kk += 32) {
        // ---- A operands: two 16x32 f16 weight tiles, contiguous k order (same as B) ---
        // lanes 0-15 : a[0..15]=K kk+0..15
        // lanes 16-31: a[0..15]=K kk+16..31
        v16h a[2];
#pragma unroll
        for (int u = 0; u < 2; ++u) {
            const _Float16* wrow =
                Wh + (size_t)(m0 + u * 16 + l) * Cin + kk + half16 * 16;
            const v8h w0 = *(const v8h*)(wrow);       // global_load_b128
            const v8h w1 = *(const v8h*)(wrow + 8);   // global_load_b128
#pragma unroll
            for (int i = 0; i < 8; ++i) { a[u][i] = w0[i]; a[u][8 + i] = w1[i]; }
        }

        // ---- per-K-step BN transform coefficients (shared by 4 n-tiles) --
        const int cb = kk + half16 * 16;
        float sc16[16], sh16[16], lo16[16];
#pragma unroll
        for (int i = 0; i < 16; ++i) {
            const int c = cb + i;
            sc16[i] = tsc[c]; sh16[i] = tsh[c]; lo16[i] = tlo[c];
        }

        // ---- 4 n-tiles: build B (32x16, lanes<16 K=kk+i, lanes>=16
        //      K=kk+16+i) with fused BN+ReLU, feed both m-tiles -------------
#pragma unroll
        for (int t = 0; t < 4; ++t) {
            const int n = p0 + t * 16 + l;
            v16h bv;
#pragma unroll
            for (int i = 0; i < 16; ++i) {
                float v = (float)X[(size_t)(cb + i) * P_ + n];
                v = fmaxf(v * sc16[i] + sh16[i], lo16[i]);
                bv[i] = (_Float16)v;
            }
            acc[0][t] = __builtin_amdgcn_wmma_f32_16x16x32_f16(
                false, a[0], false, bv, (short)0, acc[0][t], false, false);
            acc[1][t] = __builtin_amdgcn_wmma_f32_16x16x32_f16(
                false, a[1], false, bv, (short)0, acc[1][t], false, false);
        }
    }

    // ---- epilogue: +bias, BN stats (half-wave shfl reduce + atomics), store
#pragma unroll
    for (int u = 0; u < 2; ++u) {
#pragma unroll
        for (int r = 0; r < 8; ++r) {
            const int ch = m0 + u * 16 + r + half16 * 8;  // C/D row mapping
            const float bch = bias[ch];
            float s = 0.0f, s2 = 0.0f;
#pragma unroll
            for (int t = 0; t < 4; ++t) {
                const float y = acc[u][t][r] + bch;
                s += y; s2 += y * y;
                Y[(size_t)ch * P_ + (p0 + t * 16 + l)] = (_Float16)y;
            }
#pragma unroll
            for (int off = 1; off < 16; off <<= 1) { // stays in 16-lane half
                s  += __shfl_xor(s,  off, 32);
                s2 += __shfl_xor(s2, off, 32);
            }
            if (l == 0) {
                atomicAdd(&stats[ch], s);
                atomicAdd(&stats[Cout + ch], s2);
            }
        }
    }
}

// ---------------------------------------------------------------------------
// 5) BN finalize: stats -> (scale, shift, lo=0) for the NEXT layer's fused load
// ---------------------------------------------------------------------------
__global__ void rcri_finalize_kernel(const float* __restrict__ stats,
                                     const float* __restrict__ gamma,
                                     const float* __restrict__ beta,
                                     float* __restrict__ tr_out,
                                     int C, int Cstride, float invP)
{
    int c = threadIdx.x;
    if (c >= C) return;
    const float mean = stats[c] * invP;
    const float var  = stats[C + c] * invP - mean * mean;
    const float sc   = gamma[c] * rsqrtf(var + BN_EPS_);
    tr_out[c]               = sc;
    tr_out[Cstride + c]     = beta[c] - mean * sc;
    tr_out[2 * Cstride + c] = 0.0f;   // ReLU floor
}

// ---------------------------------------------------------------------------
// 6) final: BN4 + ReLU + max over K -> out[B][256][S]  (row read as 4x b128)
// ---------------------------------------------------------------------------
__global__ void rcri_maxk_kernel(const _Float16* __restrict__ Y4,  // [256][P]
                                 const float* __restrict__ tr,     // stride 256
                                 float* __restrict__ outp)
{
    const int bs = blockIdx.x * blockDim.x + threadIdx.x;   // b*S + s
    const int c  = blockIdx.y;
    if (bs >= B_ * S_) return;
    const float sc = tr[c], sh = tr[256 + c];
    const v8h* row = (const v8h*)(Y4 + (size_t)c * P_ + (size_t)bs * K_);
    float m = -INFINITY;
#pragma unroll
    for (int j = 0; j < 4; ++j) {
        const v8h r8 = row[j];
#pragma unroll
        for (int k = 0; k < 8; ++k)
            m = fmaxf(m, fmaxf((float)r8[k] * sc + sh, 0.0f));
    }
    const int b = bs >> 11, s = bs & (S_ - 1);
    outp[(size_t)b * 256 * S_ + (size_t)c * S_ + s] = m;
}

// ---------------------------------------------------------------------------
// launch
// ---------------------------------------------------------------------------
extern "C" void kernel_launch(void* const* d_in, const int* in_sizes, int n_in,
                              void* d_out, int out_size, void* d_ws, size_t ws_size,
                              hipStream_t stream)
{
    const float* contour = (const float*)d_in[0];
    const float* LOA     = (const float*)d_in[1];
    const float* fmaps   = (const float*)d_in[2];
    const float* fe_w1   = (const float*)d_in[3];
    const float* fe_b1   = (const float*)d_in[4];
    const float* fe_g1   = (const float*)d_in[5];
    const float* fe_be1  = (const float*)d_in[6];
    const float* fe_w2   = (const float*)d_in[7];
    const float* fe_b2   = (const float*)d_in[8];
    const float* fe_g2   = (const float*)d_in[9];
    const float* fe_be2  = (const float*)d_in[10];
    const float* fl_w1   = (const float*)d_in[11];
    const float* fl_b1   = (const float*)d_in[12];
    const float* fl_g1   = (const float*)d_in[13];
    const float* fl_be1  = (const float*)d_in[14];
    const float* fl_w2   = (const float*)d_in[15];
    const float* fl_b2   = (const float*)d_in[16];
    const float* fl_g2   = (const float*)d_in[17];
    const float* fl_be2  = (const float*)d_in[18];

    float* out         = (float*)d_out;
    float* out_sampled = out;                       // B*S*3
    float* out_loa     = out + B_ * S_ * 3;         // B*S*3
    float* out_feat    = out + 2 * B_ * S_ * 3;     // B*256*S

    // ---- carve workspace (~605 MB) ------------------------------------
    char* ws = (char*)d_ws;
    size_t off = 0;
    auto carve = [&](size_t bytes) -> char* {
        char* p = ws + off;
        off = (off + bytes + 255) & ~(size_t)255;
        return p;
    };
    _Float16* x1   = (_Float16*)carve((size_t)32  * P_ * 2);  // rif (padded)
    _Float16* y1   = (_Float16*)carve((size_t)32  * P_ * 2);
    _Float16* xin3 = (_Float16*)carve((size_t)128 * P_ * 2);  // [y2 | grouped]
    _Float16* y3   = (_Float16*)carve((size_t)128 * P_ * 2);
    _Float16* y4   = (_Float16*)carve((size_t)256 * P_ * 2);
    _Float16* W1h  = (_Float16*)carve(32  * 32  * 2);
    _Float16* W2h  = (_Float16*)carve(64  * 32  * 2);
    _Float16* W3h  = (_Float16*)carve(128 * 128 * 2);
    _Float16* W4h  = (_Float16*)carve(256 * 128 * 2);
    float* stats   = (float*)carve(960 * 4);   // [64][128][256][512]
    float* stats1  = stats;
    float* stats2  = stats + 64;
    float* stats3  = stats + 192;
    float* stats4  = stats + 448;
    float* t1 = (float*)carve(96  * 4);
    float* t2 = (float*)carve(96  * 4);
    float* t3 = (float*)carve(384 * 4);
    float* t4 = (float*)carve(384 * 4);
    float* t5 = (float*)carve(768 * 4);

    const float invP = 1.0f / (float)P_;

    // 0) init stats + identity transforms; zero-fill padded rif channels
    rcri_init_kernel<<<1, 256, 0, stream>>>(stats, 960, t1, t3);
    {
        const long long n8 = (long long)24 * P_ / 8;   // rows 8..31, v8h units
        rcri_zeropad_kernel<<<(unsigned)((n8 + 255) / 256), 256, 0, stream>>>(
            (v8h*)(x1 + (size_t)8 * P_), n8);
    }

    // 1) weights -> f16 (Cin zero-padded to %32)
    rcri_wconv_kernel<<<(32 * 32 + 255) / 256, 256, 0, stream>>>(fe_w1, W1h, 32, 8, 32);
    rcri_wconv_kernel<<<(64 * 32 + 255) / 256, 256, 0, stream>>>(fe_w2, W2h, 64, 32, 32);
    rcri_wconv_kernel<<<(128 * 128 + 255) / 256, 256, 0, stream>>>(fl_w1, W3h, 128, 128, 128);
    rcri_wconv_kernel<<<(256 * 128 + 255) / 256, 256, 0, stream>>>(fl_w2, W4h, 256, 128, 128);

    // 2) geometry features + sampled/new_LOA outputs
    rcri_geom_kernel<<<P_ / 256, 256, 0, stream>>>(contour, LOA, x1, out_sampled, out_loa);

    // 3) gather feature_maps into concat half of layer-3 input
    rcri_gather_kernel<<<dim3(P_ / 256, D_), 256, 0, stream>>>(fmaps, xin3);

    // GEMM grid: block = 4 waves; wave = 32(M) x 64(N) tile; 256 pos/block
    // 4) L1: 32 <- 32(pad of 8)
    rcri_gemm_bn_kernel<<<dim3(P_ / 256, 32 / 32), 128, 0, stream>>>(
        x1, W1h, fe_b1, t1, stats1, y1, 32, 32);
    rcri_finalize_kernel<<<1, 32, 0, stream>>>(stats1, fe_g1, fe_be1, t2, 32, 32, invP);

    // 5) L2: 64 <- 32 ; writes rows [0..63] of xin3
    rcri_gemm_bn_kernel<<<dim3(P_ / 256, 64 / 32), 128, 0, stream>>>(
        y1, W2h, fe_b2, t2, stats2, xin3, 32, 64);
    rcri_finalize_kernel<<<1, 64, 0, stream>>>(stats2, fe_g2, fe_be2, t3, 64, 128, invP);

    // 6) L3: 128 <- 128 (concat; upper 64 channels raw via identity transform)
    rcri_gemm_bn_kernel<<<dim3(P_ / 256, 128 / 32), 128, 0, stream>>>(
        xin3, W3h, fl_b1, t3, stats3, y3, 128, 128);
    rcri_finalize_kernel<<<1, 128, 0, stream>>>(stats3, fl_g1, fl_be1, t4, 128, 128, invP);

    // 7) L4: 256 <- 128
    rcri_gemm_bn_kernel<<<dim3(P_ / 256, 256 / 32), 128, 0, stream>>>(
        y3, W4h, fl_b2, t4, stats4, y4, 128, 256);
    rcri_finalize_kernel<<<1, 256, 0, stream>>>(stats4, fl_g2, fl_be2, t5, 256, 256, invP);

    // 8) BN4 + ReLU + max over K -> out
    rcri_maxk_kernel<<<dim3((B_ * S_) / 256, 256), 256, 0, stream>>>(y4, t5, out_feat);
}
